// Decoder_89584427860122
// MI455X (gfx1250) — matrix operand order repaired and checked by reading, not compiled, not run on hardware
//
#include <hip/hip_runtime.h>
#include <math.h>

typedef __bf16 bf16;
typedef __attribute__((ext_vector_type(16))) bf16 v16bf;
typedef __attribute__((ext_vector_type(8)))  bf16 v8bf;
typedef __attribute__((ext_vector_type(8)))  float v8f;

#define B_   32
#define T_   400
#define U_   1024
#define G_   4096
#define PRE_ 256
#define NM_  80
#define NF_  161   // 160 frames + 1 stop
#define ZO_  0.1f
#define HLDP 1032  // padded LDS row stride for h (1024 + 8 bf16 -> bank rotation)

// ---------------------------------------------------------------------------
// Weight packing: B-matrix fragments for V_WMMA_F32_16X16X32_BF16.
// Packed layout: for blk in [0,64), kstep in [0,K/32), nt in [0,4), lane, i:
//   offset  = (((blk*ksteps + kstep)*4 + nt)*32 + lane)*16 + i
//   k       = kstep*32 + (i>>3)*16 + (lane>>4)*8 + (i&7)
//   origcol = nt*1024 + blk*16 + (lane&15)     (gate-interleaved columns)
// ---------------------------------------------------------------------------
__global__ void pack_b_kernel(const float* __restrict__ W, bf16* __restrict__ out,
                              int ksteps) {
    int idx = blockIdx.x * 256 + threadIdx.x;     // K*4096 elements total
    int i    = idx & 15;
    int lane = (idx >> 4) & 31;
    int nt   = (idx >> 9) & 3;
    int rest = idx >> 11;
    int kstep = rest % ksteps;
    int blk   = rest / ksteps;
    int k = kstep * 32 + (i >> 3) * 16 + (lane >> 4) * 8 + (i & 7);
    int origcol = nt * 1024 + blk * 16 + (lane & 15);
    out[idx] = (bf16)W[(long)k * G_ + origcol];
}

// gp[j*4096 + blk*64 + nt*16 + n] = param_j[nt*1024 + blk*16 + n]
__global__ void pack_params_kernel(const float* a, const float* b1, const float* b2,
                                   const float* bias, float* __restrict__ out) {
    int pc = blockIdx.x * 256 + threadIdx.x;      // 0..4095
    int n    = pc & 15;
    int nt   = (pc >> 4) & 3;
    int blk  = pc >> 6;
    int oc = nt * 1024 + blk * 16 + n;
    out[pc]           = a[oc];
    out[4096 + pc]    = b1[oc];
    out[8192 + pc]    = b2[oc];
    out[12288 + pc]   = bias[oc];
}

__global__ void zero_state_kernel(bf16* h0, float* c1, float* c2) {
    int idx = blockIdx.x * 256 + threadIdx.x;     // 32768
    h0[idx] = (bf16)0.0f;
    c1[idx] = 0.0f;
    c2[idx] = 0.0f;
}

// ---------------------------------------------------------------------------
// Prenet: p[t][b][:] = relu(relu(x[b][t][:]@pw1+pb1)@pw2+pb2), bf16 out
// ---------------------------------------------------------------------------
__global__ __launch_bounds__(256) void prenet_kernel(
    const float* __restrict__ x,
    const float* __restrict__ pw1, const float* __restrict__ pb1,
    const float* __restrict__ pw2, const float* __restrict__ pb2,
    bf16* __restrict__ p) {
    __shared__ float xrow[NM_];
    __shared__ float mid[PRE_];
    int t = blockIdx.x >> 5;          // 0..399
    int b = blockIdx.x & 31;
    int j = threadIdx.x;
    if (j < NM_) xrow[j] = x[((long)b * T_ + t) * NM_ + j];
    __syncthreads();
    float acc = pb1[j];
    #pragma unroll 4
    for (int k = 0; k < NM_; ++k) acc += xrow[k] * pw1[k * PRE_ + j];
    mid[j] = fmaxf(acc, 0.0f);
    __syncthreads();
    float acc2 = pb2[j];
    #pragma unroll 4
    for (int k = 0; k < PRE_; ++k) acc2 += mid[k] * pw2[k * PRE_ + j];
    p[(long)blockIdx.x * PRE_ + j] = (bf16)fmaxf(acc2, 0.0f);
}

// ---------------------------------------------------------------------------
// CDNA5 async global->LDS copy (ASYNCcnt) + A-fragment load helpers
// ---------------------------------------------------------------------------
__device__ __forceinline__ unsigned lds_addr(const void* p) {
    return (unsigned)(unsigned long long)p;   // low 32 bits = LDS offset
}
__device__ __forceinline__ void async_b128(unsigned lds, const bf16* g) {
    asm volatile("global_load_async_to_lds_b128 %0, %1, off"
                 :: "v"(lds), "v"(g) : "memory");
}
__device__ __forceinline__ void wait_async0() {
    asm volatile("s_wait_asynccnt 0x0" ::: "memory");
}
__device__ __forceinline__ v16bf ldA(const bf16* __restrict__ p) {
    v8bf lo = *(const v8bf*)p;          // K = half*8 + 0..7
    v8bf hi = *(const v8bf*)(p + 16);   // K = half*8 + 16..23
    v16bf r;
    #pragma unroll
    for (int i = 0; i < 8; ++i) { r[i] = lo[i]; r[i + 8] = hi[i]; }
    return r;
}
__device__ __forceinline__ float sigf(float v) { return 1.0f / (1.0f + __expf(-v)); }

// ---------------------------------------------------------------------------
// One MI-LSTM timestep. 64 blocks x 256 threads (8 waves).
// Block blk owns packed gate-columns [blk*64, blk*64+64) -> hidden units
// [blk*16, blk*16+16), all 4 gates. Wave w: mt=w&1 (batch tile), nt=w>>1 (gate).
// XKS = x-input ksteps (8 for layer1, 32 for layer2); XLDP = padded LDS stride.
// ---------------------------------------------------------------------------
template<int XKS, int XLDP>
__global__ __launch_bounds__(256) void mi_lstm_step_kernel(
    const bf16* __restrict__ xin,                        // per-t input [32, XKS*32]
    const bf16* __restrict__ hprev,                      // [32, 1024]
    const bf16* __restrict__ wxp, const bf16* __restrict__ whp,
    const float* __restrict__ gp,                        // packed alpha,b1,b2,bias
    float* __restrict__ c, bf16* __restrict__ hout) {
    constexpr int XK = XKS * 32;
    __shared__ bf16  hbuf[32 * HLDP];
    __shared__ bf16  xbuf[32 * XLDP];
    __shared__ float zbuf[32 * 64];
    const int blk  = blockIdx.x;
    const int tid  = threadIdx.x;
    const int wv   = tid >> 5, lane = tid & 31;
    const int mt   = wv & 1,  nt   = wv >> 1;
    const int half = lane >> 4, nn = lane & 15;
    const int row  = mt * 16 + nn;                       // batch row

    // -- async-stage A operands into padded LDS (ASYNCcnt path) --
    #pragma unroll
    for (int q = tid; q < 32 * 128; q += 256) {          // hprev: 64 KB
        int r = q >> 7, c8 = q & 127;
        async_b128(lds_addr(&hbuf[r * HLDP + c8 * 8]), hprev + r * U_ + c8 * 8);
    }
    #pragma unroll
    for (int q = tid; q < 32 * (XK / 8); q += 256) {     // x_t
        int r = q / (XK / 8), c8 = q % (XK / 8);
        async_b128(lds_addr(&xbuf[r * XLDP + c8 * 8]), xin + r * XK + c8 * 8);
    }
    wait_async0();
    __syncthreads();

    // -- interleaved WMMA: h@wh (32 ksteps) and x@wx (XKS ksteps) --
    const bf16* aH = &hbuf[row * HLDP + half * 8];
    const bf16* aX = &xbuf[row * XLDP + half * 8];
    const bf16* bH = whp + ((((long)blk * 32) * 4 + nt) * 32 + lane) * 16;
    const bf16* bX = wxp + ((((long)blk * XKS) * 4 + nt) * 32 + lane) * 16;
    v8f acc_hw = {}, acc_xw = {};
    #pragma unroll 4
    for (int ks = 0; ks < XKS; ++ks) {
        __builtin_prefetch(bH + (long)(ks + 4) * 2048, 0, 1);
        v16bf a  = ldA(aH + ks * 32);
        v16bf b  = *(const v16bf*)(bH + (long)ks * 2048);
        acc_hw = __builtin_amdgcn_wmma_f32_16x16x32_bf16(false, a, false, b,
                                                         (short)0, acc_hw, false, false);
        v16bf a2 = ldA(aX + ks * 32);
        v16bf b2 = *(const v16bf*)(bX + (long)ks * 2048);
        acc_xw = __builtin_amdgcn_wmma_f32_16x16x32_bf16(false, a2, false, b2,
                                                         (short)0, acc_xw, false, false);
    }
    #pragma unroll 4
    for (int ks = XKS; ks < 32; ++ks) {
        __builtin_prefetch(bH + (long)(ks + 4) * 2048, 0, 1);
        v16bf a = ldA(aH + ks * 32);
        v16bf b = *(const v16bf*)(bH + (long)ks * 2048);
        acc_hw = __builtin_amdgcn_wmma_f32_16x16x32_bf16(false, a, false, b,
                                                         (short)0, acc_hw, false, false);
    }

    // -- MI combine per element, stash z in LDS --
    const int pcb = blk * 64 + nt * 16 + nn;
    float al = gp[pcb], bb1 = gp[4096 + pcb], bb2 = gp[8192 + pcb], bi = gp[12288 + pcb];
    #pragma unroll
    for (int r = 0; r < 8; ++r) {
        float hw = acc_hw[r], xw = acc_xw[r];
        int m = mt * 16 + half * 8 + r;
        zbuf[m * 64 + nt * 16 + nn] = al * xw * hw + bb1 * xw + bb2 * hw + bi;
    }
    __syncthreads();

    // -- gate fusion + zoneout: 512 (b,u) pairs, 2 per thread --
    #pragma unroll
    for (int it = 0; it < 2; ++it) {
        int idx = tid + it * 256;
        int b = idx >> 4, u = idx & 15;
        float iv = zbuf[b * 64 + u];
        float fv = zbuf[b * 64 + 16 + u];
        float gv = zbuf[b * 64 + 32 + u];
        float ov = zbuf[b * 64 + 48 + u];
        int unit = blk * 16 + u;
        float c_old = c[b * U_ + unit];
        float h_old = (float)hbuf[b * HLDP + unit];
        float c_new = sigf(fv) * c_old + sigf(iv) * tanhf(gv);
        float h_new = sigf(ov) * tanhf(c_new);
        c[b * U_ + unit] = ZO_ * c_old + (1.0f - ZO_) * c_new;
        hout[b * U_ + unit] = (bf16)(ZO_ * h_old + (1.0f - ZO_) * h_new);
    }
}

// ---------------------------------------------------------------------------
// Final: out[b][t][0:160] = h2_t[b]@fw + fb ; out[..][160] = sigmoid(h2@sw+sb)
// ---------------------------------------------------------------------------
__global__ __launch_bounds__(256) void head_kernel(
    const bf16* __restrict__ h2,          // [T][B][U]
    const float* __restrict__ sw, const float* __restrict__ sb,
    const float* __restrict__ fw, const float* __restrict__ fb,
    float* __restrict__ out) {
    __shared__ float hrow[U_];
    int t = blockIdx.x >> 5;
    int b = blockIdx.x & 31;
    int j = threadIdx.x;
    const bf16* hp = h2 + (long)blockIdx.x * U_;
    #pragma unroll
    for (int i = 0; i < 4; ++i) hrow[j * 4 + i] = (float)hp[j * 4 + i];
    __syncthreads();
    if (j < 160) {
        float acc = fb[j];
        #pragma unroll 4
        for (int k = 0; k < U_; ++k) acc += hrow[k] * fw[k * 160 + j];
        out[((long)b * T_ + t) * NF_ + j] = acc;
    } else if (j == 160) {
        float acc = sb[0];
        #pragma unroll 4
        for (int k = 0; k < U_; ++k) acc += hrow[k] * sw[k];
        out[((long)b * T_ + t) * NF_ + 160] = 1.0f / (1.0f + __expf(-acc));
    }
}

// ---------------------------------------------------------------------------
extern "C" void kernel_launch(void* const* d_in, const int* in_sizes, int n_in,
                              void* d_out, int out_size, void* d_ws, size_t ws_size,
                              hipStream_t stream) {
    const float* x    = (const float*)d_in[0];
    const float* pw1  = (const float*)d_in[1];
    const float* pb1  = (const float*)d_in[2];
    const float* pw2  = (const float*)d_in[3];
    const float* pb2  = (const float*)d_in[4];
    const float* wx1  = (const float*)d_in[5];
    const float* wh1  = (const float*)d_in[6];
    const float* a1   = (const float*)d_in[7];
    const float* b11  = (const float*)d_in[8];
    const float* b21  = (const float*)d_in[9];
    const float* bs1  = (const float*)d_in[10];
    const float* wx2  = (const float*)d_in[11];
    const float* wh2  = (const float*)d_in[12];
    const float* a2   = (const float*)d_in[13];
    const float* b12  = (const float*)d_in[14];
    const float* b22  = (const float*)d_in[15];
    const float* bs2  = (const float*)d_in[16];
    const float* sw   = (const float*)d_in[17];
    const float* sb   = (const float*)d_in[18];
    const float* fw   = (const float*)d_in[19];
    const float* fb   = (const float*)d_in[20];
    float* out = (float*)d_out;

    char* w = (char*)d_ws;
    size_t off = 0;
    auto alloc = [&](size_t bytes) { char* p = w + off; off += (bytes + 255) & ~255ull; return p; };
    bf16*  p    = (bf16*) alloc((size_t)T_ * B_ * PRE_ * 2);
    bf16*  hs1  = (bf16*) alloc((size_t)T_ * B_ * U_ * 2);
    bf16*  hs2  = (bf16*) alloc((size_t)T_ * B_ * U_ * 2);
    bf16*  h0   = (bf16*) alloc((size_t)B_ * U_ * 2);
    bf16*  whp1 = (bf16*) alloc((size_t)U_ * G_ * 2);
    bf16*  whp2 = (bf16*) alloc((size_t)U_ * G_ * 2);
    bf16*  wxp1 = (bf16*) alloc((size_t)PRE_ * G_ * 2);
    bf16*  wxp2 = (bf16*) alloc((size_t)U_ * G_ * 2);
    float* gp1  = (float*)alloc((size_t)4 * G_ * 4);
    float* gp2  = (float*)alloc((size_t)4 * G_ * 4);
    float* c1   = (float*)alloc((size_t)B_ * U_ * 4);
    float* c2   = (float*)alloc((size_t)B_ * U_ * 4);

    // Pack weights into WMMA B-fragment layout (gate-interleaved), bf16.
    pack_b_kernel<<<(U_ * G_) / 256, 256, 0, stream>>>(wh1, whp1, 32);
    pack_b_kernel<<<(U_ * G_) / 256, 256, 0, stream>>>(wh2, whp2, 32);
    pack_b_kernel<<<(PRE_ * G_) / 256, 256, 0, stream>>>(wx1, wxp1, 8);
    pack_b_kernel<<<(U_ * G_) / 256, 256, 0, stream>>>(wx2, wxp2, 32);
    pack_params_kernel<<<G_ / 256, 256, 0, stream>>>(a1, b11, b21, bs1, gp1);
    pack_params_kernel<<<G_ / 256, 256, 0, stream>>>(a2, b12, b22, bs2, gp2);
    zero_state_kernel<<<(B_ * U_) / 256, 256, 0, stream>>>(h0, c1, c2);

    prenet_kernel<<<T_ * B_, 256, 0, stream>>>(x, pw1, pb1, pw2, pb2, p);

    // Layer 1 recurrence (400 sequential step kernels, graph-captured).
    for (int t = 0; t < T_; ++t) {
        const bf16* hp = (t == 0) ? h0 : (hs1 + (long)(t - 1) * B_ * U_);
        mi_lstm_step_kernel<8, 264><<<64, 256, 0, stream>>>(
            p + (long)t * B_ * PRE_, hp, wxp1, whp1, gp1,
            c1, hs1 + (long)t * B_ * U_);
    }
    // Layer 2 recurrence.
    for (int t = 0; t < T_; ++t) {
        const bf16* hp = (t == 0) ? h0 : (hs2 + (long)(t - 1) * B_ * U_);
        mi_lstm_step_kernel<32, HLDP><<<64, 256, 0, stream>>>(
            hs1 + (long)t * B_ * U_, hp, wxp2, whp2, gp2,
            c2, hs2 + (long)t * B_ * U_);
    }

    head_kernel<<<T_ * B_, 256, 0, stream>>>(hs2, sw, sb, fw, fb, out);
}
